// Attention_50182397886756
// MI455X (gfx1250) — matrix operand order repaired and checked by reading, not compiled, not run on hardware
//
#include <hip/hip_runtime.h>
#include <hip/hip_bf16.h>
#include <math.h>

// Problem constants (match reference)
#define BATCH 8
#define TLEN  1024
#define CDIM  1024
#define NHEAD 16
#define HDIM  64
#define MROWS (BATCH * TLEN)   // 8192

typedef __attribute__((ext_vector_type(16))) __bf16 v16bf;
typedef __attribute__((ext_vector_type(8)))  __bf16 v8bf;
typedef __attribute__((ext_vector_type(8)))  float  v8f;
typedef __attribute__((ext_vector_type(4)))  unsigned int u32x4;
typedef __attribute__((ext_vector_type(8)))  unsigned int u32x8;

static __device__ __forceinline__ v8f wmma_bf16(v16bf a, v16bf b, v8f c) {
  // D = A(16x32 bf16) * B(32x16 bf16) + C(16x16 f32)
  return __builtin_amdgcn_wmma_f32_16x16x32_bf16(
      /*neg_a=*/false, a, /*neg_b=*/false, b,
      /*c_mod=*/(short)0, c, /*reuse_a=*/false, /*reuse_b=*/false);
}

static __device__ __forceinline__ v16bf cat8(v8bf lo, v8bf hi) {
  return __builtin_shufflevector(lo, hi, 0, 1, 2, 3, 4, 5, 6, 7,
                                 8, 9, 10, 11, 12, 13, 14, 15);
}

// ---------------------------------------------------------------------------
// Kernel 1: qkv = x @ W_attn^T + b_attn   (M=8192, N=3072, K=1024)
// fp32 in -> bf16 LDS tiles -> WMMA f32 accum.
// Workgroup = 256 threads (8 waves), block tile 128(M) x 128(N), K-step 32.
// Wave tile 32x64 = 2x4 WMMA tiles (8 WMMA per K-step per wave).
// Q/K/V selection is block-uniform (tile never straddles a 1024 boundary).
// ---------------------------------------------------------------------------
__global__ __launch_bounds__(256)
void qkv_gemm(const float* __restrict__ x, const float* __restrict__ W,
              const float* __restrict__ bias,
              __bf16* __restrict__ Qb, __bf16* __restrict__ Kb,
              __bf16* __restrict__ Vb) {
  __shared__ __bf16 As[128 * 32];
  __shared__ __bf16 Bs[128 * 32];

  const int tid  = threadIdx.x;
  const int lane = tid & 31;
  const int wave = tid >> 5;
  const int wm   = wave & 3;   // 4 waves along M (32 rows each)
  const int wn   = wave >> 2;  // 2 waves along N (64 cols each)
  const int m0   = blockIdx.y * 128;
  const int n0   = blockIdx.x * 128;

  v8f acc[2][4] = {};

  const int ldrow = tid >> 1;        // 0..127
  const int ldcol = (tid & 1) * 16;  // 0 or 16

  const int kb1  = (lane < 16) ? 0 : 8;   // A-fragment K base
  const int kbB  = (lane < 16) ? 0 : 16;  // unused; B fragments also use kb1
  const int mrow = lane & 15;

  for (int kk = 0; kk < CDIM; kk += 32) {
    // Stage A tile 128x32 and B tile 128x32 (fp32 -> bf16)
    const float4* ap =
        (const float4*)(x + (size_t)(m0 + ldrow) * CDIM + kk + ldcol);
    const float4* bp =
        (const float4*)(W + (size_t)(n0 + ldrow) * CDIM + kk + ldcol);
#pragma unroll
    for (int i = 0; i < 4; ++i) {
      float4 fa = ap[i];
      __bf16* da = &As[ldrow * 32 + ldcol + i * 4];
      da[0] = (__bf16)fa.x; da[1] = (__bf16)fa.y;
      da[2] = (__bf16)fa.z; da[3] = (__bf16)fa.w;
      float4 fb = bp[i];
      __bf16* db = &Bs[ldrow * 32 + ldcol + i * 4];
      db[0] = (__bf16)fb.x; db[1] = (__bf16)fb.y;
      db[2] = (__bf16)fb.z; db[3] = (__bf16)fb.w;
    }
    __syncthreads();

    v16bf a[2], b[4];
#pragma unroll
    for (int mt = 0; mt < 2; ++mt) {
      int r = wm * 32 + mt * 16 + mrow;
      v8bf lo = *(const v8bf*)&As[r * 32 + kb1];
      v8bf hi = *(const v8bf*)&As[r * 32 + kb1 + 16];
      a[mt] = cat8(lo, hi);
    }
#pragma unroll
    for (int nt = 0; nt < 4; ++nt) {
      int r = wn * 64 + nt * 16 + mrow;
      b[nt] = cat8(*(const v8bf*)&Bs[r * 32 + kb1], *(const v8bf*)&Bs[r * 32 + kb1 + 16]);
    }
#pragma unroll
    for (int mt = 0; mt < 2; ++mt)
#pragma unroll
      for (int nt = 0; nt < 4; ++nt)
        acc[mt][nt] = wmma_bf16(a[mt], b[nt], acc[mt][nt]);
    __syncthreads();
  }

  // Epilogue: uniform Q/K/V select, bias, head layout [B, nh, T, hd], bf16.
  const int which = n0 >> 10;  // block-uniform: 0=Q 1=K 2=V
  __bf16* __restrict__ dst = (which == 0) ? Qb : (which == 1) ? Kb : Vb;
  const float qscale = (which == 0) ? 0.125f : 1.0f;  // 1/sqrt(64) into Q

  const int rbase = (lane < 16) ? 0 : 8;
#pragma unroll
  for (int mt = 0; mt < 2; ++mt) {
#pragma unroll
    for (int nt = 0; nt < 4; ++nt) {
#pragma unroll
      for (int r = 0; r < 8; ++r) {
        int gm = m0 + wm * 32 + mt * 16 + rbase + r;
        int gn = n0 + wn * 64 + nt * 16 + mrow;
        float v = (acc[mt][nt][r] + bias[gn]) * qscale;
        int bidx = gm >> 10;          // / TLEN
        int t    = gm & (TLEN - 1);
        int c    = gn & (CDIM - 1);
        int h = c >> 6, d = c & (HDIM - 1);
        size_t off = (((size_t)bidx * NHEAD + h) * TLEN + t) * HDIM + d;
        dst[off] = (__bf16)v;
      }
    }
  }
}

// ---------------------------------------------------------------------------
// Kernel 2: causal flash attention, one wave per (b, h, 32-row q-tile).
// K/V tiles DMA'd into LDS with the Tensor Data Mover (tensor_load_to_lds,
// TENSORcnt-tracked); S = Q K^T via WMMA; online softmax with shfl_xor over
// 16-lane C-layout groups; P -> LDS -> A-layout; O += P V via WMMA.
// Single wave => same-wave LDS ops are in-order, no workgroup barriers.
// ---------------------------------------------------------------------------
__global__ __launch_bounds__(32)
void flash_attn(const __bf16* __restrict__ Qb, const __bf16* __restrict__ Kb,
                const __bf16* __restrict__ Vb, __bf16* __restrict__ AO) {
  __shared__ __bf16 Ks[32 * 64];  // [kv][ch]
  __shared__ __bf16 Vs[32 * 64];  // [kv][d] (row-major; B-frags gathered)
  __shared__ __bf16 Ps[32 * 32];  // probabilities [q][kv]

  const int lane = threadIdx.x;
  const int q0 = blockIdx.x * 32;
  const int h  = blockIdx.y;
  const int bb = blockIdx.z;

  const __bf16* qbase = Qb + ((size_t)bb * NHEAD + h) * TLEN * HDIM;
  const __bf16* kbase = Kb + ((size_t)bb * NHEAD + h) * TLEN * HDIM;
  const __bf16* vbase = Vb + ((size_t)bb * NHEAD + h) * TLEN * HDIM;

  const int kb1   = (lane < 16) ? 0 : 8;
  const int kbB   = (lane < 16) ? 0 : 16;
  const int mrow  = lane & 15;
  const int rbase = (lane < 16) ? 0 : 8;

  // Q fragments: 2 M-tiles x 2 K-steps (64 channels), kept in registers.
  v16bf qa[2][2];
#pragma unroll
  for (int mt = 0; mt < 2; ++mt) {
    const __bf16* qrow = qbase + (size_t)(q0 + mt * 16 + mrow) * HDIM;
#pragma unroll
    for (int ks = 0; ks < 2; ++ks) {
      v8bf lo = *(const v8bf*)(qrow + ks * 32 + kb1);
      v8bf hi = *(const v8bf*)(qrow + ks * 32 + kb1 + 16);
      qa[mt][ks] = cat8(lo, hi);
    }
  }

  v8f o[2][4] = {};
  float mi[2][8], li[2][8];
#pragma unroll
  for (int mt = 0; mt < 2; ++mt)
#pragma unroll
    for (int r = 0; r < 8; ++r) { mi[mt][r] = -3.0e38f; li[mt][r] = 0.0f; }

  // TDM descriptor Group1 (geometry, same for K and V):
  // data_size=2B; tensor_dim0=64; tensor_dim1=1024; tile 64x32; stride0=64.
  u32x8 g1;
  g1[0] = 0x00010000u;                    // data_size code 1 (2 bytes)
  g1[1] = ((unsigned)HDIM) << 16;         // tensor_dim0[15:0]
  g1[2] = ((unsigned)TLEN & 0xffffu) << 16;  // dim0 hi=0 | tensor_dim1[15:0]
  g1[3] = ((unsigned)HDIM) << 16;         // dim1 hi=0 | tile_dim0=64
  g1[4] = 32u;                            // tile_dim1=32, tile_dim2=0
  g1[5] = (unsigned)HDIM;                 // tensor_dim0_stride lo32 = 64
  g1[6] = 0u;
  g1[7] = 0u;

  const unsigned lds_ks = (unsigned)(size_t)(void*)&Ks[0];
  const unsigned lds_vs = (unsigned)(size_t)(void*)&Vs[0];

  const int nk = (q0 >> 5) + 1;  // causal: kv tiles with k0 <= q0
  for (int kt = 0; kt < nk; ++kt) {
    const int k0 = kt * 32;

    // ---- TDM: DMA K and V 32x64 bf16 tiles from HBM into LDS ----
    unsigned long long gk = (unsigned long long)(size_t)(kbase + (size_t)k0 * HDIM);
    unsigned long long gv = (unsigned long long)(size_t)(vbase + (size_t)k0 * HDIM);
    u32x4 g0k, g0v;
    g0k[0] = 1u;  g0k[1] = lds_ks;
    g0k[2] = (unsigned)gk;
    g0k[3] = ((unsigned)(gk >> 32) & 0x01FFFFFFu) | 0x80000000u;  // type=2
    g0v[0] = 1u;  g0v[1] = lds_vs;
    g0v[2] = (unsigned)gv;
    g0v[3] = ((unsigned)(gv >> 32) & 0x01FFFFFFu) | 0x80000000u;
    asm volatile(
        "s_wait_dscnt 0x0\n\t"        // old LDS readers must drain first
        "tensor_load_to_lds %0, %1\n\t"
        "tensor_load_to_lds %2, %3\n\t"
        "s_wait_tensorcnt 0x0"
        :
        : "s"(g0k), "s"(g1), "s"(g0v), "s"(g1)
        : "memory");

    // S = Q K^T  (32x32 fp32 in C layout)
    v8f s[2][2] = {};
#pragma unroll
    for (int ks = 0; ks < 2; ++ks) {
      v16bf bk[2];
#pragma unroll
      for (int nt = 0; nt < 2; ++nt)
        bk[nt] = cat8(*(const v8bf*)&Ks[(nt * 16 + mrow) * 64 + ks * 32 + kb1], *(const v8bf*)&Ks[(nt * 16 + mrow) * 64 + ks * 32 + kb1 + 16]);
#pragma unroll
      for (int mt = 0; mt < 2; ++mt)
#pragma unroll
        for (int nt = 0; nt < 2; ++nt)
          s[mt][nt] = wmma_bf16(qa[mt][ks], bk[nt], s[mt][nt]);
    }

    // Causal mask (diagonal tile only).
    if (k0 == q0) {
#pragma unroll
      for (int mt = 0; mt < 2; ++mt)
#pragma unroll
        for (int nt = 0; nt < 2; ++nt)
#pragma unroll
          for (int r = 0; r < 8; ++r) {
            int qrow = mt * 16 + rbase + r;
            int kcol = nt * 16 + mrow;
            if (kcol > qrow) s[mt][nt][r] = -3.0e38f;
          }
    }

    // Online softmax (rows live in 16-lane groups; xor masks < 16 stay
    // inside the group on wave32).
#pragma unroll
    for (int mt = 0; mt < 2; ++mt) {
#pragma unroll
      for (int r = 0; r < 8; ++r) {
        float v = fmaxf(s[mt][0][r], s[mt][1][r]);
#pragma unroll
        for (int msk = 8; msk >= 1; msk >>= 1)
          v = fmaxf(v, __shfl_xor(v, msk, 32));
        float newm  = fmaxf(mi[mt][r], v);
        float alpha = __expf(mi[mt][r] - newm);
        mi[mt][r] = newm;
        float p0 = __expf(s[mt][0][r] - newm);
        float p1 = __expf(s[mt][1][r] - newm);
        s[mt][0][r] = p0; s[mt][1][r] = p1;
        float rs = p0 + p1;
#pragma unroll
        for (int msk = 8; msk >= 1; msk >>= 1)
          rs += __shfl_xor(rs, msk, 32);
        li[mt][r] = li[mt][r] * alpha + rs;
#pragma unroll
        for (int nv = 0; nv < 4; ++nv) o[mt][nv][r] *= alpha;
      }
    }

    // P (C layout) -> LDS -> A layout. Single wave: DS ops are in-order.
#pragma unroll
    for (int mt = 0; mt < 2; ++mt)
#pragma unroll
      for (int nt = 0; nt < 2; ++nt)
#pragma unroll
        for (int r = 0; r < 8; ++r)
          Ps[(mt * 16 + rbase + r) * 32 + nt * 16 + mrow] =
              (__bf16)s[mt][nt][r];

    v16bf pa[2];
#pragma unroll
    for (int mt = 0; mt < 2; ++mt) {
      const __bf16* prow = &Ps[(mt * 16 + mrow) * 32];
      v8bf lo = *(const v8bf*)(prow + kb1);
      v8bf hi = *(const v8bf*)(prow + kb1 + 16);
      pa[mt] = cat8(lo, hi);
    }
    // V B-fragments gathered from row-major Vs: element e -> kv = kb1+(e&7)+16*(e>>3).
#pragma unroll
    for (int nv = 0; nv < 4; ++nv) {
      v16bf vb;
      const int d = nv * 16 + mrow;
#pragma unroll
      for (int e = 0; e < 16; ++e)
        vb[e] = Vs[(kb1 + (e & 7) + ((e >> 3) << 4)) * 64 + d];
#pragma unroll
      for (int mt = 0; mt < 2; ++mt)
        o[mt][nv] = wmma_bf16(pa[mt], vb, o[mt][nv]);
    }
  }

  // Epilogue: normalize and write bf16 to [B, T, C] (heads re-interleaved).
  __bf16* obase = AO + (size_t)bb * TLEN * CDIM + h * HDIM;
#pragma unroll
  for (int mt = 0; mt < 2; ++mt) {
    float inv[8];
#pragma unroll
    for (int r = 0; r < 8; ++r) inv[r] = 1.0f / li[mt][r];
#pragma unroll
    for (int nv = 0; nv < 4; ++nv)
#pragma unroll
      for (int r = 0; r < 8; ++r) {
        int qrow = q0 + mt * 16 + rbase + r;
        int d = nv * 16 + mrow;
        obase[(size_t)qrow * CDIM + d] = (__bf16)(o[mt][nv][r] * inv[r]);
      }
  }
}

// ---------------------------------------------------------------------------
// Kernel 3: out = AO(bf16) @ W_proj^T + b_proj, fp32 output.
// Same tiling as kernel 1 (128x128 block, 32x64 wave tile); A already bf16.
// ---------------------------------------------------------------------------
__global__ __launch_bounds__(256)
void proj_gemm(const __bf16* __restrict__ A, const float* __restrict__ W,
               const float* __restrict__ bias, float* __restrict__ out) {
  __shared__ __bf16 As[128 * 32];
  __shared__ __bf16 Bs[128 * 32];

  const int tid  = threadIdx.x;
  const int lane = tid & 31;
  const int wave = tid >> 5;
  const int wm   = wave & 3;
  const int wn   = wave >> 2;
  const int m0   = blockIdx.y * 128;
  const int n0   = blockIdx.x * 128;

  v8f acc[2][4] = {};

  const int ldrow = tid >> 1;
  const int ldcol = (tid & 1) * 16;

  const int kb1  = (lane < 16) ? 0 : 8;
  const int kbB  = (lane < 16) ? 0 : 16;
  const int mrow = lane & 15;

  for (int kk = 0; kk < CDIM; kk += 32) {
    *(v16bf*)&As[ldrow * 32 + ldcol] =
        *(const v16bf*)(A + (size_t)(m0 + ldrow) * CDIM + kk + ldcol);
    const float4* bp =
        (const float4*)(W + (size_t)(n0 + ldrow) * CDIM + kk + ldcol);
#pragma unroll
    for (int i = 0; i < 4; ++i) {
      float4 f = bp[i];
      __bf16* d = &Bs[ldrow * 32 + ldcol + i * 4];
      d[0] = (__bf16)f.x; d[1] = (__bf16)f.y;
      d[2] = (__bf16)f.z; d[3] = (__bf16)f.w;
    }
    __syncthreads();

    v16bf a[2], b[4];
#pragma unroll
    for (int mt = 0; mt < 2; ++mt) {
      int r = wm * 32 + mt * 16 + mrow;
      v8bf lo = *(const v8bf*)&As[r * 32 + kb1];
      v8bf hi = *(const v8bf*)&As[r * 32 + kb1 + 16];
      a[mt] = cat8(lo, hi);
    }
#pragma unroll
    for (int nt = 0; nt < 4; ++nt) {
      int r = wn * 64 + nt * 16 + mrow;
      b[nt] = cat8(*(const v8bf*)&Bs[r * 32 + kb1], *(const v8bf*)&Bs[r * 32 + kb1 + 16]);
    }
#pragma unroll
    for (int mt = 0; mt < 2; ++mt)
#pragma unroll
      for (int nt = 0; nt < 4; ++nt)
        acc[mt][nt] = wmma_bf16(a[mt], b[nt], acc[mt][nt]);
    __syncthreads();
  }

  const int rbase = (lane < 16) ? 0 : 8;
#pragma unroll
  for (int mt = 0; mt < 2; ++mt)
#pragma unroll
    for (int nt = 0; nt < 4; ++nt)
#pragma unroll
      for (int r = 0; r < 8; ++r) {
        int gm = m0 + wm * 32 + mt * 16 + rbase + r;
        int gn = n0 + wn * 64 + nt * 16 + mrow;
        out[(size_t)gm * CDIM + gn] = acc[mt][nt][r] + bias[gn];
      }
}

// ---------------------------------------------------------------------------
extern "C" void kernel_launch(void* const* d_in, const int* in_sizes, int n_in,
                              void* d_out, int out_size, void* d_ws,
                              size_t ws_size, hipStream_t stream) {
  (void)in_sizes; (void)n_in; (void)out_size; (void)ws_size;
  const float* x      = (const float*)d_in[0];
  const float* W_attn = (const float*)d_in[1];
  const float* b_attn = (const float*)d_in[2];
  const float* W_proj = (const float*)d_in[3];
  const float* b_proj = (const float*)d_in[4];
  float* out = (float*)d_out;

  // Workspace layout: Q | K | V | AO, each bf16 of B*nh*T*hd (= B*T*C) elems.
  char* ws = (char*)d_ws;
  const size_t seg = (size_t)BATCH * NHEAD * TLEN * HDIM * sizeof(__bf16);
  __bf16* Qb = (__bf16*)(ws + 0 * seg);
  __bf16* Kb = (__bf16*)(ws + 1 * seg);
  __bf16* Vb = (__bf16*)(ws + 2 * seg);
  __bf16* AO = (__bf16*)(ws + 3 * seg);

  dim3 g1(3 * CDIM / 128, MROWS / 128);  // (24, 64)
  qkv_gemm<<<g1, 256, 0, stream>>>(x, W_attn, b_attn, Qb, Kb, Vb);

  dim3 g2(TLEN / 32, NHEAD, BATCH);      // (32, 16, 8)
  flash_attn<<<g2, 32, 0, stream>>>(Qb, Kb, Vb, AO);

  dim3 g3(CDIM / 128, MROWS / 128);      // (8, 64)
  proj_gemm<<<g3, 256, 0, stream>>>(AO, W_proj, b_proj, out);
}
